// QNetwork_89094801588550
// MI455X (gfx1250) — compile-verified, matrix operand order checked by reading, not run on hardware
//
#include <hip/hip_runtime.h>
#include <hip/hip_bf16.h>
#include <math.h>

typedef __attribute__((ext_vector_type(16))) __bf16 v16bf;
typedef __attribute__((ext_vector_type(8)))  __bf16 v8bf;
typedef __attribute__((ext_vector_type(8)))  float  v8f;
typedef __attribute__((ext_vector_type(4)))  float  v4f;

// ---------- helpers ----------

__device__ __forceinline__ __bf16 f2b(float f) {
  // round-to-nearest-even fp32 -> bf16
  unsigned u = __float_as_uint(f);
  unsigned r = (u + 0x7FFFu + ((u >> 16) & 1u)) >> 16;
  unsigned short s = (unsigned short)r;
  __bf16 b;
  __builtin_memcpy(&b, &s, 2);
  return b;
}

__device__ __forceinline__ v8f wmma_bf16(v16bf a, v16bf b, v8f c) {
  return __builtin_amdgcn_wmma_f32_16x16x32_bf16(false, a, false, b, (short)0, c,
                                                 false, false);
}

// A-fragment (16x32 bf16, ISA 7.12.2 layout) from a row-major 64-wide bf16 row.
// lane 0-15: this lane is matrix row M=lane, holds K = [kh*32 .. ] bytes 0-7,16-23
// lane 16-31: same row set, holds K offsets +8 / +24.
__device__ __forceinline__ v16bf load_a16(const __bf16* row, int lane, int khalf) {
  int kbase = ((lane & 16) ? 8 : 0) + khalf * 32;
  union { v16bf v; v8bf h[2]; } u;
  u.h[0] = *(const v8bf*)(row + kbase);
  u.h[1] = *(const v8bf*)(row + kbase + 16);
  return u.v;
}

__device__ __forceinline__ v16bf load_a16_f32(const float* row, int lane, int khalf) {
  int kbase = ((lane & 16) ? 8 : 0) + khalf * 32;
  union { v16bf v; __bf16 e[16]; } u;
  v4f f0 = *(const v4f*)(row + kbase);
  v4f f1 = *(const v4f*)(row + kbase + 4);
  v4f f2 = *(const v4f*)(row + kbase + 16);
  v4f f3 = *(const v4f*)(row + kbase + 20);
#pragma unroll
  for (int i = 0; i < 4; ++i) {
    u.e[i]      = f2b(f0[i]);
    u.e[4 + i]  = f2b(f1[i]);
    u.e[8 + i]  = f2b(f2[i]);
    u.e[12 + i] = f2b(f3[i]);
  }
  return u.v;
}

// ---------- kernel 1: build h = concat(x0, ann); zero agg; bf16 mirror ----------

__global__ void build_h_kernel(const float* __restrict__ x0,
                               const float* __restrict__ ann,
                               float* __restrict__ h_f, __bf16* __restrict__ h_b,
                               float* __restrict__ agg, int N) {
  int idx = blockIdx.x * 256 + threadIdx.x;      // N*64 total
  if (idx >= N * 64) return;
  int node = idx >> 6, j = idx & 63;
  float v = (j < 62) ? x0[node * 62 + j] : ann[node * 2 + (j - 62)];
  h_f[idx] = v;
  h_b[idx] = f2b(v);
  agg[idx] = 0.0f;
}

// ---------- kernel 2: fp32 -> bf16 weight conversion ----------

__global__ void f2bf_kernel(const float* __restrict__ in, __bf16* __restrict__ out,
                            int n) {
  int i = blockIdx.x * 256 + threadIdx.x;
  if (i < n) out[i] = f2b(in[i]);
}

// ---------- kernel 3: edge messages m = h[src] @ W_msg^T + b; atomic scatter to agg ----------
// one wave = 16 edges; K=64 via two 16x16x32 bf16 WMMAs per 16-col tile.

__global__ void msg_scatter_kernel(const __bf16* __restrict__ h_b,
                                   const int* __restrict__ edges,
                                   const __bf16* __restrict__ wmsg_b,
                                   const float* __restrict__ b_msg,
                                   float* __restrict__ agg, int E) {
  int lane = threadIdx.x & 31;
  int wave = blockIdx.x * (blockDim.x >> 5) + (threadIdx.x >> 5);
  int e0 = wave * 16;
  if (e0 >= E) return;
  int m = lane & 15;
  int mbase = (lane & 16) ? 8 : 0;

  int src = edges[2 * (e0 + m) + 0];
  const __bf16* rowA = h_b + (long)src * 64;
  v16bf A0 = load_a16(rowA, lane, 0);
  v16bf A1 = load_a16(rowA, lane, 1);

  int dsts[8];
#pragma unroll
  for (int i = 0; i < 8; ++i) dsts[i] = edges[2 * (e0 + i + mbase) + 1];

#pragma unroll
  for (int t = 0; t < 4; ++t) {
    int col = t * 16 + m;                       // output column this lane carries
    const __bf16* rowB = wmsg_b + (long)col * 64;  // W_msg row = B column (m = h @ W^T)
    v16bf B0 = load_a16(rowB, lane, 0);
    v16bf B1 = load_a16(rowB, lane, 1);
    v8f c = {};
    c = wmma_bf16(A0, B0, c);
    c = wmma_bf16(A1, B1, c);
    float bias = b_msg[col];
#pragma unroll
    for (int i = 0; i < 8; ++i) {
      // D layout: VGPR i -> row (i + mbase), col = lane&15 within tile
      atomicAdd(&agg[(long)dsts[i] * 64 + col], c[i] + bias);
    }
  }
}

// ---------- kernel 4: GRU update ----------
// one wave = 16 nodes. gx = agg@Wih^T+bih, gh = h@Whh^T+bhh (192 cols = 12 tiles).
// r,z staged in wave-private LDS; n-gate + blend fused.

__global__ void gru_kernel(const float* __restrict__ agg,
                           const float* __restrict__ h_f,
                           const __bf16* __restrict__ h_b,
                           const __bf16* __restrict__ wih_b,
                           const __bf16* __restrict__ whh_b,
                           const float* __restrict__ bih,
                           const float* __restrict__ bhh,
                           float* __restrict__ hn_f, __bf16* __restrict__ hn_b,
                           int N) {
  __shared__ float lds_rz[8][2][16 * 64];  // 64 KB / block
  int lane = threadIdx.x & 31;
  int w = threadIdx.x >> 5;
  int node0 = (blockIdx.x * 8 + w) * 16;
  if (node0 >= N) return;
  int m = lane & 15;
  int mbase = (lane & 16) ? 8 : 0;

  const float* arow = agg + (long)(node0 + m) * 64;
  const __bf16* hrow = h_b + (long)(node0 + m) * 64;
  v16bf Aa0 = load_a16_f32(arow, lane, 0);
  v16bf Aa1 = load_a16_f32(arow, lane, 1);
  v16bf Ah0 = load_a16(hrow, lane, 0);
  v16bf Ah1 = load_a16(hrow, lane, 1);

  float* lr = lds_rz[w][0];
  float* lz = lds_rz[w][1];

#pragma unroll
  for (int g = 0; g < 2; ++g) {  // g=0: r gate, g=1: z gate
    float* dstg = (g == 0) ? lr : lz;
#pragma unroll
    for (int t = 0; t < 4; ++t) {
      int col = g * 64 + t * 16 + m;
      const __bf16* bi = wih_b + (long)col * 64;
      const __bf16* bh = whh_b + (long)col * 64;
      v16bf Bi0 = load_a16(bi, lane, 0), Bi1 = load_a16(bi, lane, 1);
      v16bf Bh0 = load_a16(bh, lane, 0), Bh1 = load_a16(bh, lane, 1);
      v8f cx = {}, ch = {};
      cx = wmma_bf16(Aa0, Bi0, cx);
      cx = wmma_bf16(Aa1, Bi1, cx);
      ch = wmma_bf16(Ah0, Bh0, ch);
      ch = wmma_bf16(Ah1, Bh1, ch);
      float bx = bih[col], bh2 = bhh[col];
#pragma unroll
      for (int i = 0; i < 8; ++i) {
        float s = (cx[i] + bx) + (ch[i] + bh2);
        dstg[(i + mbase) * 64 + t * 16 + m] = 1.0f / (1.0f + __expf(-s));
      }
    }
  }
  asm volatile("s_wait_dscnt 0x0" ::: "memory");  // cross-lane LDS visibility

#pragma unroll
  for (int t = 0; t < 4; ++t) {  // n gate + blend (cols 128..191 of gates)
    int col = 128 + t * 16 + m;
    const __bf16* bi = wih_b + (long)col * 64;
    const __bf16* bh = whh_b + (long)col * 64;
    v16bf Bi0 = load_a16(bi, lane, 0), Bi1 = load_a16(bi, lane, 1);
    v16bf Bh0 = load_a16(bh, lane, 0), Bh1 = load_a16(bh, lane, 1);
    v8f cx = {}, ch = {};
    cx = wmma_bf16(Aa0, Bi0, cx);
    cx = wmma_bf16(Aa1, Bi1, cx);
    ch = wmma_bf16(Ah0, Bh0, ch);
    ch = wmma_bf16(Ah1, Bh1, ch);
    float bx = bih[col], bh2 = bhh[col];
#pragma unroll
    for (int i = 0; i < 8; ++i) {
      int M = i + mbase;
      int j = t * 16 + m;
      float r = lr[M * 64 + j];
      float z = lz[M * 64 + j];
      float nn = tanhf((cx[i] + bx) + r * (ch[i] + bh2));
      long o = (long)(node0 + M) * 64 + j;
      float hv = (1.0f - z) * nn + z * h_f[o];
      hn_f[o] = hv;
      hn_b[o] = f2b(hv);
    }
  }
}

// ---------- kernel 5: select-head MLP (64->64->64->1) per node ----------

__global__ void select_kernel(const __bf16* __restrict__ hn_b,
                              const __bf16* __restrict__ w1b,
                              const float* __restrict__ b1,
                              const __bf16* __restrict__ w2b,
                              const float* __restrict__ b2,
                              const float* __restrict__ w3,
                              const float* __restrict__ b3,
                              float* __restrict__ out, int N) {
  __shared__ __bf16 l1s[8][16 * 64];  // 16 KB
  __shared__ float l2s[8][16 * 64];   // 32 KB
  int lane = threadIdx.x & 31;
  int w = threadIdx.x >> 5;
  int node0 = (blockIdx.x * 8 + w) * 16;
  if (node0 >= N) return;
  int m = lane & 15;
  int mbase = (lane & 16) ? 8 : 0;

  const __bf16* hrow = hn_b + (long)(node0 + m) * 64;
  v16bf A0 = load_a16(hrow, lane, 0);
  v16bf A1 = load_a16(hrow, lane, 1);

  __bf16* L1 = l1s[w];
#pragma unroll
  for (int t = 0; t < 4; ++t) {
    int col = t * 16 + m;
    const __bf16* br = w1b + (long)col * 64;
    v16bf B0 = load_a16(br, lane, 0), B1 = load_a16(br, lane, 1);
    v8f c = {};
    c = wmma_bf16(A0, B0, c);
    c = wmma_bf16(A1, B1, c);
    float bb = b1[col];
#pragma unroll
    for (int i = 0; i < 8; ++i) {
      float v = c[i] + bb;
      L1[(i + mbase) * 64 + t * 16 + m] = f2b(v > 0.0f ? v : 0.0f);
    }
  }
  asm volatile("s_wait_dscnt 0x0" ::: "memory");

  v16bf C0 = load_a16(L1 + m * 64, lane, 0);
  v16bf C1 = load_a16(L1 + m * 64, lane, 1);
  float* L2 = l2s[w];
#pragma unroll
  for (int t = 0; t < 4; ++t) {
    int col = t * 16 + m;
    const __bf16* br = w2b + (long)col * 64;
    v16bf B0 = load_a16(br, lane, 0), B1 = load_a16(br, lane, 1);
    v8f c = {};
    c = wmma_bf16(C0, B0, c);
    c = wmma_bf16(C1, B1, c);
    float bb = b2[col];
#pragma unroll
    for (int i = 0; i < 8; ++i) {
      float v = c[i] + bb;
      L2[(i + mbase) * 64 + t * 16 + m] = v > 0.0f ? v : 0.0f;
    }
  }
  asm volatile("s_wait_dscnt 0x0" ::: "memory");

  if (lane < 16) {  // final 64 -> 1 dot per node
    float acc = b3[0];
#pragma unroll
    for (int k = 0; k < 64; ++k) acc += L2[m * 64 + k] * w3[k];
    out[node0 + m] = acc;
  }
}

// ---------- kernel 6: argmax over eligible (first-occurrence ties) ----------

__global__ void argmax_kernel(const float* __restrict__ sel,
                              const int* __restrict__ eligible,
                              int* __restrict__ nidx_ws,
                              float* __restrict__ out_nidx) {
  __shared__ float bv[256];
  __shared__ int bi[256];
  int t = threadIdx.x;
  float best = -INFINITY;
  int besti = 0x7fffffff;
  for (int i = t; i < 4096; i += 256) {
    float v = sel[eligible[i]];
    if (v > best || (v == best && i < besti)) { best = v; besti = i; }
  }
  bv[t] = best;
  bi[t] = besti;
  __syncthreads();
  for (int s = 128; s > 0; s >>= 1) {
    if (t < s) {
      if (bv[t + s] > bv[t] || (bv[t + s] == bv[t] && bi[t + s] < bi[t])) {
        bv[t] = bv[t + s];
        bi[t] = bi[t + s];
      }
    }
    __syncthreads();
  }
  if (t == 0) {
    int node = eligible[bi[0]];
    *nidx_ws = node;
    *out_nidx = (float)node;
  }
}

// ---------- kernel 7: tiny head MLP on the selected node ----------

__global__ void head_kernel(const float* __restrict__ hn_f,
                            const int* __restrict__ nidx,
                            const float* __restrict__ w1, const float* __restrict__ b1,
                            const float* __restrict__ w2, const float* __restrict__ b2,
                            const float* __restrict__ w3, const float* __restrict__ b3,
                            int od, float* __restrict__ out) {
  __shared__ float hv[64], l1[64], l2[64];
  int t = threadIdx.x;
  int node = *nidx;
  if (t < 64) hv[t] = hn_f[(long)node * 64 + t];
  __syncthreads();
  if (t < 64) {
    float a = b1[t];
    for (int k = 0; k < 64; ++k) a += hv[k] * w1[t * 64 + k];
    l1[t] = a > 0.0f ? a : 0.0f;
  }
  __syncthreads();
  if (t < 64) {
    float a = b2[t];
    for (int k = 0; k < 64; ++k) a += l1[k] * w2[t * 64 + k];
    l2[t] = a > 0.0f ? a : 0.0f;
  }
  __syncthreads();
  for (int j = t; j < od; j += blockDim.x) {
    float a = b3[j];
    for (int k = 0; k < 64; ++k) a += l2[k] * w3[j * 64 + k];
    out[j] = a;
  }
}

// ---------- launch ----------

extern "C" void kernel_launch(void* const* d_in, const int* in_sizes, int n_in,
                              void* d_out, int out_size, void* d_ws, size_t ws_size,
                              hipStream_t stream) {
  const int N = 200000, E = 3200000;

  const float* x0   = (const float*)d_in[0];
  const float* ann  = (const float*)d_in[1];
  const int* edges  = (const int*)d_in[2];
  const int* elig   = (const int*)d_in[3];
  const float* Wmsg = (const float*)d_in[4];
  const float* bmsg = (const float*)d_in[5];
  const float* Wih  = (const float*)d_in[6];
  const float* Whh  = (const float*)d_in[7];
  const float* bih  = (const float*)d_in[8];
  const float* bhh  = (const float*)d_in[9];
  const float* sn_w1 = (const float*)d_in[10]; const float* sn_b1 = (const float*)d_in[11];
  const float* sn_w2 = (const float*)d_in[12]; const float* sn_b2 = (const float*)d_in[13];
  const float* sn_w3 = (const float*)d_in[14]; const float* sn_b3 = (const float*)d_in[15];
  const float* st_w1 = (const float*)d_in[16]; const float* st_b1 = (const float*)d_in[17];
  const float* st_w2 = (const float*)d_in[18]; const float* st_b2 = (const float*)d_in[19];
  const float* st_w3 = (const float*)d_in[20]; const float* st_b3 = (const float*)d_in[21];
  const float* co_w1 = (const float*)d_in[22]; const float* co_b1 = (const float*)d_in[23];
  const float* co_w2 = (const float*)d_in[24]; const float* co_b2 = (const float*)d_in[25];
  const float* co_w3 = (const float*)d_in[26]; const float* co_b3 = (const float*)d_in[27];
  const float* sp_w1 = (const float*)d_in[28]; const float* sp_b1 = (const float*)d_in[29];
  const float* sp_w2 = (const float*)d_in[30]; const float* sp_b2 = (const float*)d_in[31];
  const float* sp_w3 = (const float*)d_in[32]; const float* sp_b3 = (const float*)d_in[33];

  char* ws = (char*)d_ws;
  const size_t SZ = (size_t)N * 64;             // 12.8M elems
  float*  h_f   = (float*)(ws);                 // 51.2 MB
  float*  agg   = (float*)(ws + SZ * 4);        // 51.2 MB
  float*  hn_f  = (float*)(ws + SZ * 8);        // 51.2 MB
  __bf16* h_b   = (__bf16*)(ws + SZ * 12);      // 25.6 MB
  __bf16* hn_b  = (__bf16*)(ws + SZ * 14);      // 25.6 MB
  __bf16* wmsg_b = (__bf16*)(ws + SZ * 16);
  __bf16* wih_b  = wmsg_b + 4096;
  __bf16* whh_b  = wih_b + 12288;
  __bf16* w1_b   = whh_b + 12288;
  __bf16* w2_b   = w1_b + 4096;
  int*    nidx   = (int*)(w2_b + 4096);

  float* out = (float*)d_out;

  build_h_kernel<<<(N * 64) / 256, 256, 0, stream>>>(x0, ann, h_f, h_b, agg, N);
  f2bf_kernel<<<16, 256, 0, stream>>>(Wmsg, wmsg_b, 4096);
  f2bf_kernel<<<48, 256, 0, stream>>>(Wih, wih_b, 12288);
  f2bf_kernel<<<48, 256, 0, stream>>>(Whh, whh_b, 12288);
  f2bf_kernel<<<16, 256, 0, stream>>>(sn_w1, w1_b, 4096);
  f2bf_kernel<<<16, 256, 0, stream>>>(sn_w2, w2_b, 4096);

  msg_scatter_kernel<<<E / (16 * 8), 256, 0, stream>>>(h_b, edges, wmsg_b, bmsg,
                                                       agg, E);

  int gru_blocks = (N / 16 + 7) / 8;  // 1563
  gru_kernel<<<gru_blocks, 256, 0, stream>>>(agg, h_f, h_b, wih_b, whh_b, bih, bhh,
                                             hn_f, hn_b, N);

  select_kernel<<<gru_blocks, 256, 0, stream>>>(hn_b, w1_b, sn_b1, w2_b, sn_b2,
                                                sn_w3, sn_b3, out, N);

  argmax_kernel<<<1, 256, 0, stream>>>(out, elig, nidx, out + N);

  head_kernel<<<1, 128, 0, stream>>>(hn_f, nidx, st_w1, st_b1, st_w2, st_b2, st_w3,
                                     st_b3, 2, out + N + 1);
  head_kernel<<<1, 128, 0, stream>>>(hn_f, nidx, co_w1, co_b1, co_w2, co_b2, co_w3,
                                     co_b3, 128, out + N + 3);
  head_kernel<<<1, 128, 0, stream>>>(hn_f, nidx, sp_w1, sp_b1, sp_w2, sp_b2, sp_w3,
                                     sp_b3, 100, out + N + 131);
}
